// RNNTJoint_46170898432387
// MI455X (gfx1250) — matrix operand order repaired and checked by reading, not compiled, not run on hardware
//
#include <hip/hip_runtime.h>
#include <hip/hip_bf16.h>

// ---------------------------------------------------------------------------
// RNN-T joint:  logits[b,t,u,v] = silu(he[b,t,:] + hp[b,u,:]) . W2[:,v] + b2[v]
//   K0: W2 fp32 -> bf16 in WMMA B-fragment order (1.31 MB, L2-resident)
//   K1: he = enc @ W1[:640] + b1 ; hp = pred @ W1[640:]      (fp32, ~1 GFLOP)
//   K2: h  = silu(he ⊕ hp) materialized ONCE as bf16 in A-fragment order
//   K3: 105-GFLOP GEMM via v_wmma_f32_16x16x32_bf16; WG = 128M x 128N,
//       8 waves x 8 accum tiles, depth-2 software-pipelined B stream.
// ---------------------------------------------------------------------------

#define Bn 4
#define Tn 200
#define Un 100
#define Jn 640
#define Vn 1024
#define MTOT  (Tn * Un)         // 20000 flat (t,u) rows per batch (multiple of 16)
#define MT16  (MTOT / 16)       // 1250 16-row M-tiles per batch
#define HE_ROWS (Bn * Tn)       // 800
#define HP_ROWS (Bn * Un)       // 400
#define KT (Jn / 32)            // 20 K-steps of 32

typedef __attribute__((ext_vector_type(16))) __bf16       bf16x16;
typedef __attribute__((ext_vector_type(8)))  float        floatx8;
typedef __attribute__((ext_vector_type(8)))  unsigned int uintx8;

static __device__ __forceinline__ unsigned short f2bf(float f) {
    unsigned int u = __builtin_bit_cast(unsigned int, f);
    u += 0x7FFFu + ((u >> 16) & 1u);            // round-to-nearest-even
    return (unsigned short)(u >> 16);
}

static __device__ __forceinline__ uintx8 ld256(const unsigned short* p) {
    return *reinterpret_cast<const uintx8*>(p);
}

// ---------------------------------------------------------------------------
// K0: W2 fp32 -> bf16, pre-swizzled into WMMA B-fragment order.
// [kt][nt16][lane][16 halves]: B(32x16) lane<16 -> N=lane,K=0..15;
// lane>=16 -> N=lane-16, K=16..31.  One contiguous 32B read per lane.
// ---------------------------------------------------------------------------
__global__ __launch_bounds__(256) void w2_swizzle_kernel(
    const float* __restrict__ W2, unsigned short* __restrict__ w2s) {
    int tid = blockIdx.x * 256 + threadIdx.x;   // one thread per (kt,nt,lane)
    if (tid >= KT * 64 * 32) return;
    int lane  = tid & 31;
    int nt    = (tid >> 5) & 63;
    int kt    = tid >> 11;
    int n     = nt * 16 + (lane & 15);
    int kBase = kt * 32 + (lane >> 4) * 16;
    union { unsigned short s[16]; uint4 q[2]; } u;
#pragma unroll
    for (int e = 0; e < 16; ++e)
        u.s[e] = f2bf(W2[(size_t)(kBase + e) * Vn + n]);
    uint4* dst = reinterpret_cast<uint4*>(w2s + (size_t)tid * 16);
    dst[0] = u.q[0];
    dst[1] = u.q[1];
}

// ---------------------------------------------------------------------------
// K1: he = enc @ We + b1 ; hp = pred @ Wp  (fp32, 8-row register blocking)
// ---------------------------------------------------------------------------
__global__ __launch_bounds__(256) void proj_kernel(
    const float* __restrict__ enc, const float* __restrict__ pred,
    const float* __restrict__ W1,  const float* __restrict__ b1,
    float* __restrict__ heW, float* __restrict__ hpW) {
    __shared__ float xs[8][Jn];
    const int tid     = threadIdx.x;
    const int rowBase = blockIdx.x * 8;
    const bool isHe   = rowBase < HE_ROWS;

    const float* src = isHe ? (enc  + (size_t)rowBase * Jn)
                            : (pred + (size_t)(rowBase - HE_ROWS) * Jn);
    for (int i = tid; i < 8 * Jn; i += 256) xs[i / Jn][i % Jn] = src[i];
    __syncthreads();

    const float* Wb  = isHe ? W1 : (W1 + (size_t)Jn * Jn);
    float*       dst = isHe ? (heW + (size_t)rowBase * Jn)
                            : (hpW + (size_t)(rowBase - HE_ROWS) * Jn);
    for (int jj = 0; jj < 3; ++jj) {
        int j = tid + jj * 256;
        if (j >= Jn) break;
        float bias = isHe ? b1[j] : 0.0f;
        float acc[8];
#pragma unroll
        for (int r = 0; r < 8; ++r) acc[r] = bias;
        for (int d = 0; d < Jn; ++d) {
            float w = Wb[(size_t)d * Jn + j];
#pragma unroll
            for (int r = 0; r < 8; ++r) acc[r] = fmaf(xs[r][d], w, acc[r]);
        }
#pragma unroll
        for (int r = 0; r < 8; ++r) dst[(size_t)r * Jn + j] = acc[r];
    }
}

// ---------------------------------------------------------------------------
// K2: h = silu(he + hp) as bf16, written directly in A-fragment order:
//   hA[ ((b*1250 + mtile)*20 + kt)*32 + lane ][16 halves]
// ---------------------------------------------------------------------------
__global__ __launch_bounds__(256) void h_build_kernel(
    const float* __restrict__ heW, const float* __restrict__ hpW,
    unsigned short* __restrict__ hA) {
    const size_t tid = (size_t)blockIdx.x * 256 + threadIdx.x;  // (b,mtile,kt,lane)
    const int lane = (int)(tid & 31);
    const int kt   = (int)((tid >> 5) % KT);
    const int mt   = (int)(((tid >> 5) / KT) % MT16);
    const int b    = (int)((tid >> 5) / ((size_t)KT * MT16));

    const int m = mt * 16 + (lane & 15);
    const int t = m / Un;
    const int u = m - t * Un;
    const float* he = heW + (size_t)(b * Tn + t) * Jn;
    const float* hp = hpW + (size_t)(b * Un + u) * Jn;
    const int jBase = kt * 32 + (lane >> 4) * 16;  // 64B-aligned

    float4 e0 = *reinterpret_cast<const float4*>(he + jBase);
    float4 e1 = *reinterpret_cast<const float4*>(he + jBase + 4);
    float4 e2 = *reinterpret_cast<const float4*>(he + jBase + 8);
    float4 e3 = *reinterpret_cast<const float4*>(he + jBase + 12);
    float4 p0 = *reinterpret_cast<const float4*>(hp + jBase);
    float4 p1 = *reinterpret_cast<const float4*>(hp + jBase + 4);
    float4 p2 = *reinterpret_cast<const float4*>(hp + jBase + 8);
    float4 p3 = *reinterpret_cast<const float4*>(hp + jBase + 12);

    float x[16] = { e0.x + p0.x, e0.y + p0.y, e0.z + p0.z, e0.w + p0.w,
                    e1.x + p1.x, e1.y + p1.y, e1.z + p1.z, e1.w + p1.w,
                    e2.x + p2.x, e2.y + p2.y, e2.z + p2.z, e2.w + p2.w,
                    e3.x + p3.x, e3.y + p3.y, e3.z + p3.z, e3.w + p3.w };
    union { unsigned short s[16]; uint4 q[2]; } o;
#pragma unroll
    for (int e = 0; e < 16; ++e) {
        float s = x[e] / (1.0f + __expf(-x[e]));   // SiLU
        o.s[e] = f2bf(s);
    }
    uint4* dst = reinterpret_cast<uint4*>(hA + tid * 16);
    dst[0] = o.q[0];
    dst[1] = o.q[1];
}

// ---------------------------------------------------------------------------
// K3: GEMM via WMMA.  grid = (ceil(1250/8)=157, 8 V-blocks of 128, B=4).
// 8 waves/WG; wave w owns mtile = blockIdx.x*8 + w and the WG's full N=128
// (8 accum tiles).  All 8 waves stream identical B fragments (WGP$ dedups).
// Depth-2 manual pipeline on the flattened 160-step (kt,tile) B stream,
// A double-buffered one kt ahead.  4 waves/EU -> 120-VGPR budget.
// ---------------------------------------------------------------------------
__global__ __launch_bounds__(256)
__attribute__((amdgpu_waves_per_eu(4)))
void joint_wmma_kernel(
    const unsigned short* __restrict__ hA, const unsigned short* __restrict__ w2s,
    const float* __restrict__ b2, float* __restrict__ out) {
    const int lane  = threadIdx.x & 31;
    const int wid   = threadIdx.x >> 5;
    const int b     = blockIdx.z;
    const int nBlk  = blockIdx.y;                  // 0..7 -> 128 cols
    const int mtile = blockIdx.x * 8 + wid;
    if (mtile >= MT16) return;                     // whole-wave tail skip
    const int ntBase = nBlk * 8;

    floatx8 zero = {0.f, 0.f, 0.f, 0.f, 0.f, 0.f, 0.f, 0.f};
    floatx8 acc[8];
#pragma unroll
    for (int tI = 0; tI < 8; ++tI) acc[tI] = zero;

    // A: 512 halves (1KB) per kt, contiguous per lane.
    const unsigned short* aP =
        hA + ((size_t)(b * MT16 + mtile) * KT * 32 + lane) * 16;
    // B: step i = kt*8 + tI at bP + kt*(64*512) + tI*512 halves (imm offsets).
    const unsigned short* bP = w2s + ((size_t)ntBase * 32 + lane) * 16;
#define BOFF(i) ((size_t)((i) >> 3) * (64 * 512) + (size_t)((i) & 7) * 512)

    uintx8 aCur = ld256(aP);
    uintx8 b0   = ld256(bP + BOFF(0));
    uintx8 b1   = ld256(bP + BOFF(1));

#pragma unroll
    for (int kt = 0; kt < KT; ++kt) {
        uintx8 aNext = (kt + 1 < KT) ? ld256(aP + (size_t)(kt + 1) * 512) : aCur;
        bf16x16 aF = __builtin_bit_cast(bf16x16, aCur);
#pragma unroll
        for (int tI = 0; tI < 8; ++tI) {
            const int ni = kt * 8 + tI + 2;        // constant-folded when unrolled
            uintx8 nxt = (ni < KT * 8) ? ld256(bP + BOFF(ni)) : b0;
            bf16x16 bF = __builtin_bit_cast(bf16x16, b0);
            acc[tI] = __builtin_amdgcn_wmma_f32_16x16x32_bf16(
                false, aF, false, bF, (short)0, acc[tI], false, false);
            b0 = b1;
            b1 = nxt;
        }
        aCur = aNext;
    }
#undef BOFF

    // D layout: VGPR r -> M = r + 8*(lane>=16), N = lane&15.  No row guards:
    // 20000 % 16 == 0, invalid tiles already returned.
    const int vBase = nBlk * 128 + (lane & 15);
    const int mW    = mtile * 16 + 8 * (lane >> 4);
#pragma unroll
    for (int tI = 0; tI < 8; ++tI) {
        int v = vBase + tI * 16;
        float bias = b2[v];
        float* o = out + ((size_t)b * MTOT + mW) * Vn + v;
#pragma unroll
        for (int r = 0; r < 8; ++r)
            o[(size_t)r * Vn] = acc[tI][r] + bias;
    }
}

// ---------------------------------------------------------------------------
extern "C" void kernel_launch(void* const* d_in, const int* in_sizes, int n_in,
                              void* d_out, int out_size, void* d_ws, size_t ws_size,
                              hipStream_t stream) {
    (void)in_sizes; (void)n_in; (void)out_size; (void)ws_size;
    const float* enc  = (const float*)d_in[0];
    const float* pred = (const float*)d_in[1];
    const float* W1   = (const float*)d_in[2];
    const float* b1   = (const float*)d_in[3];
    const float* W2   = (const float*)d_in[4];
    const float* b2   = (const float*)d_in[5];
    float*       out  = (float*)d_out;

    // ws: he fp32 2.048MB | hp fp32 1.024MB | W2 bf16 1.31MB | hA bf16 102.4MB
    float*          heW = (float*)d_ws;
    float*          hpW = heW + (size_t)HE_ROWS * Jn;
    unsigned short* w2s = (unsigned short*)(hpW + (size_t)HP_ROWS * Jn);
    unsigned short* hA  = w2s + (size_t)KT * 64 * 32 * 16;

    w2_swizzle_kernel<<<dim3((KT * 64 * 32 + 255) / 256), dim3(256), 0, stream>>>(W2, w2s);
    proj_kernel<<<dim3((HE_ROWS + HP_ROWS) / 8), dim3(256), 0, stream>>>(
        enc, pred, W1, b1, heW, hpW);
    // one thread per (b, mtile, kt, lane): 4*1250*20*32 = 3.2M threads
    h_build_kernel<<<dim3(Bn * MT16 * KT * 32 / 256), dim3(256), 0, stream>>>(
        heW, hpW, hA);
    joint_wmma_kernel<<<dim3((MT16 + 7) / 8, Vn / 128, Bn), dim3(256), 0, stream>>>(
        hA, w2s, b2, out);
}
